// Bridge_43739946942807
// MI455X (gfx1250) — matrix operand order repaired and checked by reading, not compiled, not run on hardware
//
#include <hip/hip_runtime.h>
#include <hip/hip_bf16.h>
#include <math.h>

typedef __attribute__((ext_vector_type(16))) _Float16 v16h;
typedef __attribute__((ext_vector_type(8)))  float    v8f;

#define LSEQ 32768
#define DDIM 128
#define BDIM 4
#define TWIN 64
#define HDIM 8
#define MTOT (LSEQ + TWIN - 1)   // 32831 rows of xall / P
#define PSTRIDE 32               // padded row stride of P (24 valid cols)
#define BN_EPS 0.001f
#define ALPHA 0.05f

// ---------------------------------------------------------------------------
// Kernel 1: x = inputs @ w_in + b_in ; xall = concat(in_buffer, x)
// also writes out_buffer = x[-(T-1):] into the tail of d_out.
// ---------------------------------------------------------------------------
__global__ void k_proj(const float4* __restrict__ inputs4,   // (L, D) of float4 (B=4 contiguous)
                       const float*  __restrict__ inbuf,     // (63, D)
                       const float*  __restrict__ win,       // (4,)
                       const float*  __restrict__ bin,       // (1,)
                       float* __restrict__ xall,             // (MTOT, D)
                       float* __restrict__ outbuf)           // (63, D) tail of d_out
{
    int i = blockIdx.x * blockDim.x + threadIdx.x;
    if (i >= MTOT * DDIM) return;
    int r = i >> 7;          // row in xall
    int d = i & (DDIM - 1);
    if (r < TWIN - 1) {
        xall[i] = inbuf[i];
        return;
    }
    int l = r - (TWIN - 1);
    float4 v = inputs4[(size_t)l * DDIM + d];
    float x = fmaf(v.x, win[0], fmaf(v.y, win[1], fmaf(v.z, win[2], fmaf(v.w, win[3], bin[0]))));
    xall[i] = x;
    if (l >= LSEQ - (TWIN - 1))
        outbuf[(l - (LSEQ - (TWIN - 1))) * DDIM + d] = x;
}

// ---------------------------------------------------------------------------
// Kernel 1b: pre-swizzle gru_W into WMMA B-operand register layout.
// Wt[((nt*4 + c)*32 + lane)*16 + e] = f16( W[K][N] ),
//   K = c*32 + (lane>=16 ? 8 : 0) + (e<8 ? e : e+8),  N = nt*16 + (lane&15)  (0 if N>=24).
// Each lane's v16h operand is then one contiguous 32-byte load.
// ---------------------------------------------------------------------------
__global__ void k_prepW(const float* __restrict__ gruW,   // (128, 24)
                        _Float16* __restrict__ Wt)        // 2*4*32*16 halves
{
    int tid  = threadIdx.x;          // 256 threads, 1 block
    int nt   = tid >> 7;
    int c    = (tid >> 5) & 3;
    int lane = tid & 31;
    int N    = nt * 16 + (lane & 15);
    int kbas = c * 32 + ((lane >> 4) ? 8 : 0);
    _Float16* dst = Wt + (size_t)tid * 16;
    #pragma unroll
    for (int e = 0; e < 16; ++e) {
        int K = kbas + (e < 8 ? e : e + 8);
        dst[e] = (N < 24) ? (_Float16)gruW[K * 24 + N] : (_Float16)0.0f;
    }
}

// ---------------------------------------------------------------------------
// Kernel 2: P = f16(xall) @ f16(gru_W) + gru_b[0]   via v_wmma_f32_16x16x32_f16
// 256 threads = 8 waves, each wave owns one 16-row M tile. No LDS:
//  - A operand: two contiguous 8-float runs per chunk, loaded as float4s + cvt
//  - B operand: contiguous 32B v16h loads from pre-swizzled Wt (L2 broadcast)
// ---------------------------------------------------------------------------
__device__ __forceinline__ v16h make_a(const float4 f0, const float4 f1,
                                       const float4 f2, const float4 f3)
{
    v16h a;
    a[0]  = (_Float16)f0.x; a[1]  = (_Float16)f0.y;
    a[2]  = (_Float16)f0.z; a[3]  = (_Float16)f0.w;
    a[4]  = (_Float16)f1.x; a[5]  = (_Float16)f1.y;
    a[6]  = (_Float16)f1.z; a[7]  = (_Float16)f1.w;
    a[8]  = (_Float16)f2.x; a[9]  = (_Float16)f2.y;
    a[10] = (_Float16)f2.z; a[11] = (_Float16)f2.w;
    a[12] = (_Float16)f3.x; a[13] = (_Float16)f3.y;
    a[14] = (_Float16)f3.z; a[15] = (_Float16)f3.w;
    return a;
}

__global__ void k_gemm(const float* __restrict__ xall,      // (MTOT, 128)
                       const _Float16* __restrict__ Wt,     // swizzled B operands
                       const float* __restrict__ grub,      // (2, 24) flat; row0 used
                       float* __restrict__ Pg)              // (MTOT, 32)
{
    const int tid  = threadIdx.x;
    const int wave = tid >> 5;
    const int lane = tid & 31;
    const int m0   = blockIdx.x * 128 + wave * 16;

    const int mrow = lane & 15;
    const int hi   = (lane >> 4) & 1;

    // A row for this lane (clamped; rows >= MTOT only feed unstored C rows)
    int arow_i = m0 + mrow;
    if (arow_i >= MTOT) arow_i = MTOT - 1;
    const float* arow = xall + (size_t)arow_i * DDIM;

    v8f c0 = {}, c1 = {};

    #pragma unroll
    for (int c = 0; c < 4; ++c) {
        const int kb = c * 32 + hi * 8;   // start of first 8-K run for this lane
        float4 f0 = *(const float4*)(arow + kb);
        float4 f1 = *(const float4*)(arow + kb + 4);
        float4 f2 = *(const float4*)(arow + kb + 16);
        float4 f3 = *(const float4*)(arow + kb + 20);
        v16h a = make_a(f0, f1, f2, f3);

        v16h b0 = *(const v16h*)(Wt + ((size_t)(0 * 4 + c) * 32 + lane) * 16);
        v16h b1 = *(const v16h*)(Wt + ((size_t)(1 * 4 + c) * 32 + lane) * 16);

        c0 = __builtin_amdgcn_wmma_f32_16x16x32_f16(false, a, false, b0, (short)0, c0, false, false);
        c1 = __builtin_amdgcn_wmma_f32_16x16x32_f16(false, a, false, b1, (short)0, c1, false, false);
    }

    // C/D layout: VGPR r -> M = r + (lane>=16 ? 8 : 0), N = lane&15. Add gru_b[0].
    const int nco  = lane & 15;
    const int moff = hi ? 8 : 0;
    const float bias0 = grub[nco];                               // cols 0..15
    const float bias1 = (16 + nco < 24) ? grub[16 + nco] : 0.0f; // cols 16..23 (+pad)
    #pragma unroll
    for (int r = 0; r < 8; ++r) {
        int m = m0 + moff + r;
        if (m < MTOT) {
            Pg[(size_t)m * PSTRIDE + nco]      = c0[r] + bias0;
            Pg[(size_t)m * PSTRIDE + 16 + nco] = c1[r] + bias1;
        }
    }
}

// ---------------------------------------------------------------------------
// Kernel 3: GRU recurrence (1 sequence per lane, 32 per wave/block) + BN +
// leaky + BN + output projection (block-cooperative, coalesced).
// ---------------------------------------------------------------------------
__device__ __forceinline__ float sigmoid_f(float x) {
    return 1.0f / (1.0f + __expf(-x));
}

__global__ void k_gru(const float* __restrict__ Pg,    // (MTOT, 32)
                      const float* __restrict__ gruU,  // (8, 24)
                      const float* __restrict__ grub,  // (2,24); row1 used here
                      const float* __restrict__ g1, const float* __restrict__ be1,
                      const float* __restrict__ m1, const float* __restrict__ v1,
                      const float* __restrict__ g2, const float* __restrict__ be2,
                      const float* __restrict__ m2, const float* __restrict__ v2,
                      const float* __restrict__ wout,  // (8, 128)
                      const float* __restrict__ bout,  // (128,)
                      float* __restrict__ out)         // (L, 128)
{
    __shared__ float Pl[95 * 25];   // 95 rows needed, stride 25 (gcd(25,64)=1)
    __shared__ float Ul[HDIM * 24];
    __shared__ float b1s[24];
    __shared__ float ys[32 * 9];    // stride 9 to avoid bank conflicts

    const int lane = threadIdx.x;   // 32 threads = 1 wave
    const int l0   = blockIdx.x * 32;

    for (int i = lane; i < 95 * 24; i += 32) {
        int r = i / 24, k = i - r * 24;
        Pl[r * 25 + k] = Pg[(size_t)(l0 + r) * PSTRIDE + k];
    }
    for (int i = lane; i < HDIM * 24; i += 32) Ul[i] = gruU[i];
    if (lane < 24) b1s[lane] = grub[24 + lane];
    __syncthreads();

    float h[HDIM];
    #pragma unroll
    for (int i = 0; i < HDIM; ++i) h[i] = 0.0f;

    for (int t = 0; t < TWIN; ++t) {
        const float* xp = &Pl[(lane + t) * 25];
        float rec[24];
        #pragma unroll
        for (int k = 0; k < 24; ++k) rec[k] = b1s[k];
        #pragma unroll
        for (int j = 0; j < HDIM; ++j) {
            float hj = h[j];
            #pragma unroll
            for (int k = 0; k < 24; ++k)
                rec[k] = fmaf(hj, Ul[j * 24 + k], rec[k]);
        }
        #pragma unroll
        for (int i = 0; i < HDIM; ++i) {
            float z  = sigmoid_f(xp[i] + rec[i]);
            float r  = sigmoid_f(xp[HDIM + i] + rec[HDIM + i]);
            float hh = tanhf(xp[2 * HDIM + i] + r * rec[2 * HDIM + i]);
            h[i] = z * h[i] + (1.0f - z) * hh;
        }
    }

    // BN1 -> leaky -> BN2
    #pragma unroll
    for (int i = 0; i < HDIM; ++i) {
        float y = g1[i] * (h[i] - m1[i]) * rsqrtf(v1[i] + BN_EPS) + be1[i];
        y = (y > 0.0f) ? y : ALPHA * y;
        y = g2[i] * (y - m2[i]) * rsqrtf(v2[i] + BN_EPS) + be2[i];
        ys[lane * 9 + i] = y;
    }
    __syncthreads();

    // out[l, :] = y @ w_out + b_out, coalesced over d
    for (int ll = 0; ll < 32; ++ll) {
        const int l = l0 + ll;
        #pragma unroll
        for (int c = 0; c < 4; ++c) {
            int d = c * 32 + lane;
            float acc = bout[d];
            #pragma unroll
            for (int j = 0; j < HDIM; ++j)
                acc = fmaf(ys[ll * 9 + j], wout[j * DDIM + d], acc);
            out[(size_t)l * DDIM + d] = acc;
        }
    }
}

// ---------------------------------------------------------------------------
extern "C" void kernel_launch(void* const* d_in, const int* in_sizes, int n_in,
                              void* d_out, int out_size, void* d_ws, size_t ws_size,
                              hipStream_t stream) {
    const float* inputs = (const float*)d_in[0];   // (L, D, B)
    const float* inbuf  = (const float*)d_in[1];   // (63, D)
    const float* win    = (const float*)d_in[2];   // (4, 1)
    const float* bin    = (const float*)d_in[3];   // (1,)
    const float* gruW   = (const float*)d_in[4];   // (128, 24)
    const float* gruU   = (const float*)d_in[5];   // (8, 24)
    const float* grub   = (const float*)d_in[6];   // (2, 24)
    const float* g1     = (const float*)d_in[7];
    const float* be1    = (const float*)d_in[8];
    const float* m1     = (const float*)d_in[9];
    const float* v1     = (const float*)d_in[10];
    const float* g2     = (const float*)d_in[11];
    const float* be2    = (const float*)d_in[12];
    const float* m2     = (const float*)d_in[13];
    const float* v2     = (const float*)d_in[14];
    const float* wout   = (const float*)d_in[15];  // (8, 128)
    const float* bout   = (const float*)d_in[16];  // (128,)

    float* out    = (float*)d_out;                  // (L,128) then (63,128)
    float* outbuf = out + (size_t)LSEQ * DDIM;

    float* xall = (float*)d_ws;                     // (MTOT, 128) f32
    float* Pg   = xall + (size_t)MTOT * DDIM;       // (MTOT, 32)  f32
    _Float16* Wt = (_Float16*)(Pg + (size_t)MTOT * PSTRIDE); // 4096 halves

    // K1: projection + concat (independent of K1b)
    {
        int n = MTOT * DDIM;
        int blocks = (n + 255) / 256;
        k_proj<<<blocks, 256, 0, stream>>>((const float4*)inputs, inbuf, win, bin,
                                           xall, outbuf);
    }
    // K1b: swizzle gru_W into WMMA B-operand layout
    k_prepW<<<1, 256, 0, stream>>>(gruW, Wt);

    // K2: WMMA GEMM P = xall @ gru_W + gru_b[0]
    {
        int blocks = (MTOT + 127) / 128;  // 8 M-tiles of 16 per block
        k_gemm<<<blocks, 256, 0, stream>>>(xall, Wt, grub, Pg);
    }
    // K3: GRU recurrence + BN + output projection
    {
        int blocks = LSEQ / 32;
        k_gru<<<blocks, 32, 0, stream>>>(Pg, gruU, grub, g1, be1, m1, v1,
                                         g2, be2, m2, v2, wout, bout, out);
    }
}
